// EchoBlock_75892072121065
// MI455X (gfx1250) — matrix operand order repaired and checked by reading, not compiled, not run on hardware
//
#include <hip/hip_runtime.h>
#include <hip/hip_bf16.h>
#include <math.h>

// ---------------- problem constants ----------------
#define SLEN 1024
#define DMODEL 512
#define NHEADS 8
#define DHEAD 64
#define NNEUR 128
#define QKD 128   // 2*DHEAD euler feature dim

typedef __attribute__((ext_vector_type(16))) _Float16 v16h;
typedef __attribute__((ext_vector_type(8)))  _Float16 v8h;
typedef __attribute__((ext_vector_type(8)))  float    v8f;

union Frag {
    v16h v;
    v8h  h8[2];
};

// ---------------------------------------------------------------------------
// Batched GEMM via v_wmma_f32_16x16x32_f16, register-blocked 16x64 per wave.
//   C[M,N] (f32 accum, stored as OutT) = A[M,K] (f16 row-major, ld=lda)
//                                      x B[K,N] (f16 row-major, ld=ldb)
// Block = 128 threads = 4 waves; wave w computes rows m0+w*16..+15 and a
// 64-wide column strip: one A fragment is amortized over 4 B fragments and
// 4 independent accumulators (4 back-to-back independent WMMAs per K-step).
// Requires: K % 32 == 0, N % 64 == 0, M % 16 == 0.
// A-fragment layout (ISA 7.12.2, 16-bit A 16x32): lane = hi*16 + m,
//   halves [0..7]  = K[k0 + hi*8 .. +7]
//   halves [8..15] = K[k0 + 16 + hi*8 .. +7]
// B-fragment layout: lane = hi*16 + n, halves = same K order as A (column n0+j*16+n).
// ---------------------------------------------------------------------------
template <typename OutT>
__global__ void gemm_wmma(const _Float16* __restrict__ A,
                          const _Float16* __restrict__ B,
                          OutT* __restrict__ C,
                          int M, int N, int K,
                          int lda, int ldb, int ldc,
                          long long sA, long long sB, long long sC)
{
    const _Float16* Ab = A + (long long)blockIdx.z * sA;
    const _Float16* Bb = B + (long long)blockIdx.z * sB;
    OutT*           Cb = C + (long long)blockIdx.z * sC;

    const int wave = threadIdx.x >> 5;
    const int lane = threadIdx.x & 31;
    const int m0 = (blockIdx.y << 6) + (wave << 4);
    const int n0 = blockIdx.x << 6;
    if (m0 >= M || n0 >= N) return;   // uniform per wave -> EXEC stays all-ones

    const int mrow = m0 + (lane & 15);
    const int hi   = lane >> 4;

    v8f acc[4];
#pragma unroll
    for (int j = 0; j < 4; ++j)
        acc[j] = (v8f){0.f, 0.f, 0.f, 0.f, 0.f, 0.f, 0.f, 0.f};

    const _Float16* pa = Ab + (long long)mrow * lda + (hi << 3);
    const _Float16* pb = Bb + (long long)(hi << 3) * ldb + n0 + (lane & 15);
    const long long bstep = (long long)32 * ldb;

    for (int k0 = 0; k0 < K; k0 += 32) {
        Frag a;
        a.h8[0] = *(const v8h*)(pa);
        a.h8[1] = *(const v8h*)(pa + 16);
        Frag b[4];
#pragma unroll
        for (int j = 0; j < 4; ++j) {
            for (int s = 0; s < 8; ++s) {
                b[j].v[s]     = pb[(long long)s * ldb + j * 16];
                b[j].v[s + 8] = pb[(long long)(s + 16) * ldb + j * 16];
            }
        }
        __builtin_prefetch(pb + bstep, 0, 1);   // global_prefetch_b8 (streaming B)
        pa += 32;
        pb += bstep;
#pragma unroll
        for (int j = 0; j < 4; ++j)
            acc[j] = __builtin_amdgcn_wmma_f32_16x16x32_f16(
                         false, a.v, false, b[j].v, (short)0, acc[j], false, false);
    }

    const int n = lane & 15;
    OutT* pc = Cb + (long long)(m0 + (hi << 3)) * ldc + n0 + n;
#pragma unroll
    for (int j = 0; j < 4; ++j)
#pragma unroll
        for (int r = 0; r < 8; ++r)
            pc[(long long)r * ldc + j * 16] = (OutT)acc[j][r];
}

// ---------------------------------------------------------------------------
// LayerNorm over D=512, one block (256 threads) per token.
// ---------------------------------------------------------------------------
__global__ void layernorm_kernel(const float* __restrict__ x,
                                 const float* __restrict__ w,
                                 const float* __restrict__ b,
                                 float* __restrict__ nf32,
                                 _Float16* __restrict__ nf16)
{
    __shared__ float sd[256];
    const int s = blockIdx.x, t = threadIdx.x;
    const float* xr = x + (long long)s * DMODEL;
    float a0 = xr[t], a1 = xr[t + 256];
    sd[t] = a0 + a1;
    __syncthreads();
    for (int off = 128; off > 0; off >>= 1) {
        if (t < off) sd[t] += sd[t + off];
        __syncthreads();
    }
    float mu = sd[0] * (1.0f / DMODEL);
    __syncthreads();
    float d0 = a0 - mu, d1 = a1 - mu;
    sd[t] = d0 * d0 + d1 * d1;
    __syncthreads();
    for (int off = 128; off > 0; off >>= 1) {
        if (t < off) sd[t] += sd[t + off];
        __syncthreads();
    }
    float rstd = rsqrtf(sd[0] * (1.0f / DMODEL) + 1e-5f);
    float r0 = d0 * rstd * w[t] + b[t];
    float r1 = d1 * rstd * w[t + 256] + b[t + 256];
    long long base = (long long)s * DMODEL;
    nf32[base + t]       = r0;
    nf32[base + t + 256] = r1;
    nf16[base + t]       = (_Float16)r0;
    nf16[base + t + 256] = (_Float16)r1;
}

// dst[c*rows + r] = (f16) src[r*cols + c]   (dst is cols x rows, ld = rows)
__global__ void transpose_f32_to_f16(const float* __restrict__ src,
                                     _Float16* __restrict__ dst,
                                     int rows, int cols)
{
    int idx = blockIdx.x * 256 + threadIdx.x;
    if (idx >= rows * cols) return;
    int r = idx / cols, c = idx % cols;
    dst[(long long)c * rows + r] = (_Float16)src[idx];
}

// inv[i] = 1 / (1 + |W[i]|)
__global__ void invw_kernel(const float* __restrict__ W,
                            float* __restrict__ inv, int n)
{
    int i = blockIdx.x * 256 + threadIdx.x;
    if (i < n) inv[i] = 1.0f / (1.0f + fabsf(W[i]));
}

// ---------------------------------------------------------------------------
// Euler Q/K features. grid (S, H), block = DHEAD threads.
// Q layout: (S, H*128) row-major.  Kt layout: (H, 128, S) so GEMM-B rows
// (fixed feature j, varying token) are contiguous.
// ---------------------------------------------------------------------------
__global__ void qk_kernel(const float* __restrict__ normed,
                          const float* __restrict__ invwq,
                          const float* __restrict__ b_q,
                          const float* __restrict__ invwk,
                          const float* __restrict__ b_k,
                          _Float16* __restrict__ Q,
                          _Float16* __restrict__ Kt)
{
    const int s = blockIdx.x, h = blockIdx.y, d = threadIdx.x;
    const int hd = h * DHEAD + d;
    float xv = normed[(long long)s * DMODEL + hd];

    float thq = fmaf(xv, invwq[hd], b_q[hd]);
    float sq, cq;
    __sincosf(thq, &sq, &cq);
    long long qbase = (long long)s * (NHEADS * QKD) + h * QKD + d;
    Q[qbase]         = (_Float16)cq;
    Q[qbase + DHEAD] = (_Float16)sq;

    float thk = fmaf(xv, invwk[hd], b_k[hd]);
    float sk, ck;
    __sincosf(thk, &sk, &ck);
    long long kbase = ((long long)h * QKD + d) * SLEN + s;
    Kt[kbase]                           = (_Float16)ck;
    Kt[kbase + (long long)DHEAD * SLEN] = (_Float16)sk;
}

// ---------------------------------------------------------------------------
// Causal softmax over a (h,q) row of the score matrix; writes f16 probs with
// zeros above the diagonal so the following dense GEMM is exact.
// ---------------------------------------------------------------------------
__global__ void softmax_kernel(const float* __restrict__ scores,
                               _Float16* __restrict__ probs)
{
    __shared__ float sd[256];
    const int q = blockIdx.x, h = blockIdx.y, t = threadIdx.x;
    const long long base = ((long long)h * SLEN + q) * SLEN;
    const float* row = scores + base;
    _Float16* prow = probs + base;
    const float scale = 0.0883883476483184f;   // 1/sqrt(2*DHEAD)

    float v[4];
    float m = -1e30f;
#pragma unroll
    for (int i = 0; i < 4; ++i) {
        int k = t + i * 256;
        float xv = (k <= q) ? row[k] * scale : -1e30f;
        v[i] = xv;
        m = fmaxf(m, xv);
    }
    sd[t] = m; __syncthreads();
    for (int off = 128; off > 0; off >>= 1) {
        if (t < off) sd[t] = fmaxf(sd[t], sd[t + off]);
        __syncthreads();
    }
    m = sd[0]; __syncthreads();

    float sum = 0.f;
#pragma unroll
    for (int i = 0; i < 4; ++i) {
        int k = t + i * 256;
        float e = (k <= q) ? __expf(v[i] - m) : 0.f;
        v[i] = e;
        sum += e;
    }
    sd[t] = sum; __syncthreads();
    for (int off = 128; off > 0; off >>= 1) {
        if (t < off) sd[t] += sd[t + off];
        __syncthreads();
    }
    float inv = 1.0f / sd[0];
#pragma unroll
    for (int i = 0; i < 4; ++i)
        prow[t + i * 256] = (_Float16)(v[i] * inv);
}

// ---------------------------------------------------------------------------
// Resonant neuron sums: one wave per (token, neuron); 16 sincos per lane,
// wave32 shuffle reduction. Output: csin (S, 256) f16 = [cos_sum | sin_sum].
// ---------------------------------------------------------------------------
__global__ void resonant_kernel(const float* __restrict__ normed,
                                const float* __restrict__ invw,
                                const float* __restrict__ B_res,
                                _Float16* __restrict__ csin)
{
    const int s    = blockIdx.x;
    const int wave = threadIdx.x >> 5;
    const int lane = threadIdx.x & 31;
    const int n    = blockIdx.y * 8 + wave;

    const float* nr = normed + (long long)s * DMODEL;
    const float* wr = invw   + (long long)n * DMODEL;
    const float* br = B_res  + (long long)n * DMODEL;

    float cs = 0.f, sn = 0.f;
#pragma unroll 4
    for (int i = 0; i < 16; ++i) {
        int d = lane + i * 32;
        float th = fmaf(nr[d], wr[d], br[d]);
        float ss, cc;
        __sincosf(th, &ss, &cc);
        cs += cc;
        sn += ss;
    }
#pragma unroll
    for (int off = 16; off > 0; off >>= 1) {
        cs += __shfl_xor(cs, off, 32);
        sn += __shfl_xor(sn, off, 32);
    }
    if (lane == 0) {
        csin[(long long)s * 256 + n]         = (_Float16)cs;
        csin[(long long)s * 256 + NNEUR + n] = (_Float16)sn;
    }
}

// out = x + attn_scale*attn_proj + res_scale*silu(res_pre)
__global__ void final_kernel(const float* __restrict__ x,
                             const float* __restrict__ attn_proj,
                             const float* __restrict__ res_pre,
                             const float* __restrict__ attn_scale,
                             const float* __restrict__ res_scale,
                             float* __restrict__ out)
{
    int i = blockIdx.x * 256 + threadIdx.x;
    if (i >= SLEN * DMODEL) return;
    float r = res_pre[i];
    float silu = r / (1.0f + __expf(-r));
    out[i] = x[i] + attn_scale[0] * attn_proj[i] + res_scale[0] * silu;
}

// ---------------------------------------------------------------------------
extern "C" void kernel_launch(void* const* d_in, const int* in_sizes, int n_in,
                              void* d_out, int out_size, void* d_ws, size_t ws_size,
                              hipStream_t stream)
{
    const float* x          = (const float*)d_in[0];
    const float* ln_w       = (const float*)d_in[1];
    const float* ln_b       = (const float*)d_in[2];
    const float* w_q        = (const float*)d_in[3];
    const float* b_q        = (const float*)d_in[4];
    const float* w_k        = (const float*)d_in[5];
    const float* b_k        = (const float*)d_in[6];
    const float* v_w        = (const float*)d_in[7];
    const float* out_w      = (const float*)d_in[8];
    const float* W_res      = (const float*)d_in[9];
    const float* B_res      = (const float*)d_in[10];
    const float* p_cos      = (const float*)d_in[11];
    const float* p_sin      = (const float*)d_in[12];
    const float* attn_scale = (const float*)d_in[13];
    const float* res_scale  = (const float*)d_in[14];
    float* out = (float*)d_out;

    // ---- workspace carve (all offsets keep 256B alignment) ----
    char* p = (char*)d_ws;
    float*    normed_f32 = (float*)p;    p += (size_t)SLEN * DMODEL * 4;
    _Float16* normed_f16 = (_Float16*)p; p += (size_t)SLEN * DMODEL * 2;
    _Float16* vwT        = (_Float16*)p; p += (size_t)DMODEL * DMODEL * 2;
    _Float16* outwT      = (_Float16*)p; p += (size_t)DMODEL * DMODEL * 2;
    _Float16* Bcat       = (_Float16*)p; p += (size_t)256 * DMODEL * 2;  // [pcosT; psinT]
    _Float16* Qf         = (_Float16*)p; p += (size_t)SLEN * NHEADS * QKD * 2;
    _Float16* Ktf        = (_Float16*)p; p += (size_t)NHEADS * QKD * SLEN * 2;
    _Float16* Vf         = (_Float16*)p; p += (size_t)SLEN * DMODEL * 2;
    float*    scores     = (float*)p;    p += (size_t)NHEADS * SLEN * SLEN * 4;
    _Float16* probs      = (_Float16*)p; p += (size_t)NHEADS * SLEN * SLEN * 2;
    _Float16* attn_cat   = (_Float16*)p; p += (size_t)SLEN * DMODEL * 2;
    float*    attn_proj  = (float*)p;    p += (size_t)SLEN * DMODEL * 4;
    _Float16* csin       = (_Float16*)p; p += (size_t)SLEN * 256 * 2;
    float*    res_pre    = (float*)p;    p += (size_t)SLEN * DMODEL * 4;
    float*    invw_res   = (float*)p;    p += (size_t)NNEUR * DMODEL * 4;
    float*    invwq      = (float*)p;    p += (size_t)NHEADS * DHEAD * 4;
    float*    invwk      = (float*)p;    p += (size_t)NHEADS * DHEAD * 4;
    (void)ws_size; (void)in_sizes; (void)n_in; (void)out_size;

    // 1) LayerNorm
    layernorm_kernel<<<SLEN, 256, 0, stream>>>(x, ln_w, ln_b, normed_f32, normed_f16);

    // 2) weight prep
    transpose_f32_to_f16<<<(DMODEL * DMODEL + 255) / 256, 256, 0, stream>>>(v_w, vwT, DMODEL, DMODEL);
    transpose_f32_to_f16<<<(DMODEL * DMODEL + 255) / 256, 256, 0, stream>>>(out_w, outwT, DMODEL, DMODEL);
    transpose_f32_to_f16<<<(DMODEL * NNEUR + 255) / 256, 256, 0, stream>>>(p_cos, Bcat, DMODEL, NNEUR);
    transpose_f32_to_f16<<<(DMODEL * NNEUR + 255) / 256, 256, 0, stream>>>(p_sin, Bcat + (size_t)NNEUR * DMODEL, DMODEL, NNEUR);
    invw_kernel<<<(NNEUR * DMODEL + 255) / 256, 256, 0, stream>>>(W_res, invw_res, NNEUR * DMODEL);
    invw_kernel<<<(NHEADS * DHEAD + 255) / 256, 256, 0, stream>>>(w_q, invwq, NHEADS * DHEAD);
    invw_kernel<<<(NHEADS * DHEAD + 255) / 256, 256, 0, stream>>>(w_k, invwk, NHEADS * DHEAD);

    // 3) Euler Q/K features
    qk_kernel<<<dim3(SLEN, NHEADS), DHEAD, 0, stream>>>(normed_f32, invwq, b_q, invwk, b_k, Qf, Ktf);

    // 4) V = normed @ v_w^T   (f16 out)
    gemm_wmma<_Float16><<<dim3(DMODEL / 64, SLEN / 64, 1), 128, 0, stream>>>(
        normed_f16, vwT, Vf, SLEN, DMODEL, DMODEL, DMODEL, DMODEL, DMODEL, 0, 0, 0);

    // 5) scores[h] = Q_h @ K_h^T   (batched over heads, f32 out)
    gemm_wmma<float><<<dim3(SLEN / 64, SLEN / 64, NHEADS), 128, 0, stream>>>(
        Qf, Ktf, scores, SLEN, SLEN, QKD,
        NHEADS * QKD, SLEN, SLEN,
        (long long)QKD, (long long)QKD * SLEN, (long long)SLEN * SLEN);

    // 6) causal softmax -> f16 probs
    softmax_kernel<<<dim3(SLEN, NHEADS), 256, 0, stream>>>(scores, probs);

    // 7) attn_cat[:, h*64:(h+1)*64] = probs_h @ V_h  (batched, strided C)
    gemm_wmma<_Float16><<<dim3(1, SLEN / 64, NHEADS), 128, 0, stream>>>(
        probs, Vf, attn_cat, SLEN, DHEAD, SLEN,
        SLEN, DMODEL, DMODEL,
        (long long)SLEN * SLEN, (long long)DHEAD, (long long)DHEAD);

    // 8) attn_proj = attn_cat @ out_w^T
    gemm_wmma<float><<<dim3(DMODEL / 64, SLEN / 64, 1), 128, 0, stream>>>(
        attn_cat, outwT, attn_proj, SLEN, DMODEL, DMODEL, DMODEL, DMODEL, DMODEL, 0, 0, 0);

    // 9) resonant neuron sums (the transcendental-heavy part)
    resonant_kernel<<<dim3(SLEN, NNEUR / 8), 256, 0, stream>>>(normed_f32, invw_res, B_res, csin);

    // 10) res_pre = [cos|sin] @ [pcosT; psinT]
    gemm_wmma<float><<<dim3(DMODEL / 64, SLEN / 64, 1), 128, 0, stream>>>(
        csin, Bcat, res_pre, SLEN, DMODEL, 256, 256, DMODEL, DMODEL, 0, 0, 0);

    // 11) residual + scales + silu
    final_kernel<<<(SLEN * DMODEL + 255) / 256, 256, 0, stream>>>(
        x, attn_proj, res_pre, attn_scale, res_scale, out);
}
